// BBoxLayer_30614526886243
// MI455X (gfx1250) — matrix operand order repaired and checked by reading, not compiled, not run on hardware
//
#include <hip/hip_runtime.h>

// ---------------- problem constants ----------------
#define BATCH 32
#define CIN   3
#define HH    384
#define WW    384
#define CFEAT 256
#define HF    32
#define WF    32
#define MDIM  64
#define NOUT  353          // H - HF + 1
#define NROWS 35           // HF + 3 (row-shift trick, dy=0..3)
#define ROWLEN 416         // padded staged row (>= 352+15+16+15+1), 16B-multiple
#define NTILE 23           // ceil(353/16)
#define NGROUP 89          // ceil(353/4)

#define ASYNC_STAGE 1      // gfx1250 GLOBAL_LOAD_ASYNC_TO_LDS_B128 staging

typedef __attribute__((ext_vector_type(16))) __bf16 v16bf;
typedef __attribute__((ext_vector_type(8)))  __bf16 v8bf;
typedef __attribute__((ext_vector_type(8)))  float  v8f;

__device__ __forceinline__ float sigmoidf(float x) {
    return 1.0f / (1.0f + __expf(-x));
}

#if ASYNC_STAGE
// One 16-byte async chunk: global (16B-aligned) -> LDS offset (16B-aligned).
// Tracked on ASYNCcnt; bypasses VGPRs entirely (CDNA5 ISA).
__device__ __forceinline__ void async_ld16(unsigned int ldsByteOff, const void* gaddr) {
    asm volatile("global_load_async_to_lds_b128 %0, %1, off"
                 :: "v"(ldsByteOff), "v"((unsigned long long)(uintptr_t)gaddr)
                 : "memory");
}
__device__ __forceinline__ void wait_async0() {
    asm volatile("s_wait_asynccnt 0x0" ::: "memory");
}
#endif

// ---------------- utility kernels ----------------
__global__ void zero_bf16_kernel(__bf16* p, int n) {
    int i = blockIdx.x * blockDim.x + threadIdx.x;
    if (i < n) p[i] = (__bf16)0.0f;
}

__global__ void zero_f32_kernel(float* p, int n) {
    int i = blockIdx.x * blockDim.x + threadIdx.x;
    if (i < n) p[i] = 0.0f;
}

__global__ void cvt_f32_bf16_kernel(const float* __restrict__ s, __bf16* __restrict__ d, int n) {
    int i = blockIdx.x * blockDim.x + threadIdx.x;
    if (i < n) d[i] = (__bf16)s[i];
}

// ---------------- 3x3 valid conv + zero-pad(1) + sigmoid -> bf16 ----------------
__global__ void conv3x3_sig_kernel(const float* __restrict__ x,
                                   const float* __restrict__ w,    // [4][3][3][3]
                                   const float* __restrict__ bsrc, // [4]
                                   __bf16* __restrict__ out) {
    int idx = blockIdx.x * 256 + threadIdx.x;
    if (idx >= BATCH * 4 * HH * WW) return;
    int xw = idx % WW; int t = idx / WW;
    int yh = t % HH;   t /= HH;
    int o  = t % 4;    int b = t / 4;
    float v = 0.5f;
    if (yh > 0 && yh < HH - 1 && xw > 0 && xw < WW - 1) {
        float s = bsrc[o];
        #pragma unroll
        for (int ci = 0; ci < CIN; ++ci)
            #pragma unroll
            for (int r = 0; r < 3; ++r)
                #pragma unroll
                for (int c = 0; c < 3; ++c)
                    s += x[((b * CIN + ci) * HH + (yh - 1 + r)) * WW + (xw - 1 + c)]
                       * w[((o * CIN + ci) * 3 + r) * 3 + c];
        v = sigmoidf(s);
    }
    out[idx] = (__bf16)v;
}

// ---------------- dynamic weight generation (one workgroup per sample) ----------------
__global__ __launch_bounds__(256) void dynw_kernel(
    const float* __restrict__ fm,   // [B][256][1024]
    const float* __restrict__ meta, // [B][16][64]
    const float* __restrict__ mcw,  // [16][256]
    const float* __restrict__ mcb,  // [16]
    const float* __restrict__ mlw,  // [1024][64]
    const float* __restrict__ mlb,  // [1024]
    __bf16* __restrict__ wts)       // [B][16][32][32]
{
    extern __shared__ char smem[];
    float* s_cf = (float*)smem;            // 16*1024
    float* s_mf = s_cf + 16 * 1024;        // 16*1024
    float* s_w  = s_mf + 16 * 1024;        // 16*256
    float* s_m  = s_w  + 16 * 256;         // 16*64
    int b = blockIdx.x, tid = threadIdx.x;

    for (int i = tid; i < 16 * 256; i += 256) s_w[i] = mcw[i];
    for (int i = tid; i < 16 * 64;  i += 256) s_m[i] = meta[b * 16 * 64 + i];
    __syncthreads();

    { // cf = mc_w @ fm + mc_b
        float acc[16][4];
        #pragma unroll
        for (int k = 0; k < 16; ++k)
            #pragma unroll
            for (int pp = 0; pp < 4; ++pp) acc[k][pp] = 0.0f;
        for (int c = 0; c < CFEAT; ++c) {
            float xv[4];
            #pragma unroll
            for (int pp = 0; pp < 4; ++pp)
                xv[pp] = fm[(b * CFEAT + c) * 1024 + tid + pp * 256];
            #pragma unroll
            for (int k = 0; k < 16; ++k) {
                float wv = s_w[k * 256 + c];
                #pragma unroll
                for (int pp = 0; pp < 4; ++pp) acc[k][pp] += wv * xv[pp];
            }
        }
        #pragma unroll
        for (int k = 0; k < 16; ++k) {
            float bb = mcb[k];
            #pragma unroll
            for (int pp = 0; pp < 4; ++pp)
                s_cf[k * 1024 + tid + pp * 256] = acc[k][pp] + bb;
        }
    }
    { // mf = meta @ ml_w^T + ml_b
        float acc[16][4];
        #pragma unroll
        for (int k = 0; k < 16; ++k)
            #pragma unroll
            for (int pp = 0; pp < 4; ++pp) acc[k][pp] = 0.0f;
        for (int d = 0; d < MDIM; ++d) {
            #pragma unroll
            for (int pp = 0; pp < 4; ++pp) {
                int p = tid + pp * 256;
                float wv = mlw[p * MDIM + d];
                #pragma unroll
                for (int i = 0; i < 16; ++i) acc[i][pp] += s_m[i * MDIM + d] * wv;
            }
        }
        #pragma unroll
        for (int i = 0; i < 16; ++i)
            #pragma unroll
            for (int pp = 0; pp < 4; ++pp) {
                int p = tid + pp * 256;
                s_mf[i * 1024 + p] = acc[i][pp] + mlb[p];
            }
    }
    __syncthreads();

    // 16 independent 32x32x32 matmuls + sigmoid -> bf16
    for (int u = tid; u < 16 * 1024; u += 256) {
        int k = u >> 10, rem = u & 1023, i = rem >> 5, l = rem & 31;
        float s = 0.0f;
        #pragma unroll
        for (int j = 0; j < 32; ++j)
            s += s_cf[k * 1024 + i * 32 + j] * s_mf[k * 1024 + j * 32 + l];
        wts[(b * 16 + k) * 1024 + i * 32 + l] = (__bf16)sigmoidf(s);
    }
}

// ---------------- big 32x32-kernel conv via implicit GEMM on WMMA ----------------
//   M = (dy 0..3, o 0..3) = 16, N = 16 out-x per tile (23 tiles / 8 waves x 3),
//   K = (c, ky' 0..34, kx 0..31) = 140 WMMA K-steps.
// A[m,k] = W[o][c][ky'-dy][kx] (0 outside), B[k,n] = row[c][y0+ky'][xb+n+kx]
__global__ __launch_bounds__(256) void bigconv_wmma_kernel(
    const __bf16* __restrict__ X,    // [B][4][384][384]
    const __bf16* __restrict__ Wsrc, // [.][16][32][32], per-sample stride wstride
    const float*  __restrict__ bias, // [4] (read only when out_is_f32)
    __bf16* __restrict__ out_bf,     // interior at (+15,+15)
    float*  __restrict__ out_f,
    int wstride, int out_is_f32)
{
    extern __shared__ char smem[];
    __bf16* wl   = (__bf16*)smem;        // 16*32*32 bf16 = 32 KB, LDS off 0
    __bf16* rows = wl + 16 * 1024;       // 4*35*416 bf16,  LDS off 32768

    int b = blockIdx.x, g = blockIdx.y, tid = threadIdx.x;
    int y0 = g * 4;
    int nValidRows = (HH - y0 < NROWS) ? (HH - y0) : NROWS;   // rows with gr<384

#if ASYNC_STAGE
    // ---- weights: 2048 x 16B async chunks (both sides 16B aligned)
    for (int i = tid; i < 2048; i += 256)
        async_ld16((unsigned int)(i * 16), Wsrc + (size_t)b * wstride + i * 8);
    // ---- row interiors: per (c, valid r): 48 x 16B chunks cover x 0..383
    int totalChunks = 4 * nValidRows * 48;
    for (int i = tid; i < totalChunks; i += 256) {
        int c  = i / (nValidRows * 48);
        int rem = i - c * (nValidRows * 48);
        int r  = rem / 48;
        int ck = rem - r * 48;
        unsigned int ldsOff = 32768u + (unsigned int)(((c * NROWS + r) * ROWLEN + ck * 8) * 2);
        const __bf16* gsrc = X + ((size_t)(b * 4 + c) * HH + (y0 + r)) * WW + ck * 8;
        async_ld16(ldsOff, gsrc);
    }
    // ---- zero the ROWLEN tails (x 384..415) and fully-invalid rows with plain stores
    for (int i = tid; i < 4 * NROWS * 32; i += 256) {
        int c = i / (NROWS * 32); int rem = i - c * (NROWS * 32);
        int r = rem / 32; int xx = 384 + (rem & 31);
        rows[(c * NROWS + r) * ROWLEN + xx] = (__bf16)0.0f;
    }
    for (int r = nValidRows; r < NROWS; ++r)
        for (int i = tid; i < 4 * 384; i += 256) {
            int c = i / 384; int xx = i - c * 384;
            rows[(c * NROWS + r) * ROWLEN + xx] = (__bf16)0.0f;
        }
    wait_async0();          // my async chunks landed in LDS
    __syncthreads();        // everyone's landed
#else
    for (int i = tid; i < 16 * 1024; i += 256) wl[i] = Wsrc[(size_t)b * wstride + i];
    for (int i = tid; i < 4 * NROWS * ROWLEN; i += 256) {
        int c = i / (NROWS * ROWLEN); int rem = i % (NROWS * ROWLEN);
        int r = rem / ROWLEN; int xx = rem % ROWLEN;
        int gr = y0 + r;
        __bf16 v = (__bf16)0.0f;
        if (gr < HH && xx < WW) v = X[((size_t)(b * 4 + c) * HH + gr) * WW + xx];
        rows[i] = v;
    }
    __syncthreads();
#endif

    int wave = tid >> 5, lane = tid & 31;
    int nn = lane & 15, kh = lane >> 4;       // kh: K-half select per WMMA layout
    int dy = nn >> 2, o = nn & 3;             // A row m = lane&15 -> (dy,o)
    int xb0 = wave * 16, xb1 = (wave + 8) * 16, xb2 = (wave + 16) * 16;

    v8f acc0 = {}, acc1 = {}, acc2 = {};

    for (int c = 0; c < 4; ++c) {
        for (int kyp = 0; kyp < NROWS; ++kyp) {
            // ---- A tile (weights); contiguous K order (same as B): lanes0-15 K=0..15; hi K=16..31
            v16bf a;
            int ky = kyp - dy;
            if (ky >= 0 && ky < HF) {
                const __bf16* wp = &wl[((o * 4 + c) * HF + ky) * WF + kh * 16];
                v8bf lo = *(const v8bf*)wp;          // kx = kh*16 + 0..7   (16B aligned)
                v8bf hi = *(const v8bf*)(wp + 8);    // kx = kh*16 + 8..15
                #pragma unroll
                for (int t = 0; t < 8; ++t) { a[t] = lo[t]; a[8 + t] = hi[t]; }
            } else {
                #pragma unroll
                for (int t = 0; t < 16; ++t) a[t] = (__bf16)0.0f;
            }
            // ---- B: lane holds col n, K = kh*16 + 0..15 -> one contiguous 32B window.
            // 2-byte aligned in general: LDS runs in UNALIGNED mode under ROCm, so the
            // two ds_load_b128 this lowers to are legal (ISA §9.5 mode 3).
            const __bf16* rp = &rows[(c * NROWS + kyp) * ROWLEN + nn + kh * 16];
            v16bf bv0 = *(const v16bf*)(rp + xb0);
            acc0 = __builtin_amdgcn_wmma_f32_16x16x32_bf16(false, a, false, bv0,
                                                           (short)0, acc0, false, false);
            v16bf bv1 = *(const v16bf*)(rp + xb1);
            acc1 = __builtin_amdgcn_wmma_f32_16x16x32_bf16(false, a, false, bv1,
                                                           (short)0, acc1, false, false);
            v16bf bv2 = *(const v16bf*)(rp + xb2);   // may exceed NOUT; store guarded
            acc2 = __builtin_amdgcn_wmma_f32_16x16x32_bf16(false, a, false, bv2,
                                                           (short)0, acc2, false, false);
        }
    }

    // ---- store: D layout: lane n = lane&15, element r -> m = kh*8 + r
    #pragma unroll
    for (int j = 0; j < 3; ++j) {
        int xb = (j == 0) ? xb0 : (j == 1) ? xb1 : xb2;
        int x = xb + nn;
        if (x < NOUT) {
            #pragma unroll
            for (int r = 0; r < 8; ++r) {
                int mm = kh * 8 + r;
                int ddy = mm >> 2, oo = mm & 3;
                int y = y0 + ddy;
                if (y < NOUT) {
                    float val = (j == 0) ? acc0[r] : (j == 1) ? acc1[r] : acc2[r];
                    long oidx = ((long)(b * 4 + oo) * HH + (15 + y)) * WW + (15 + x);
                    if (out_is_f32) out_f[oidx] = val + bias[oo];
                    else            out_bf[oidx] = (__bf16)val;
                }
            }
        }
    }
}

// ---------------- host-side launcher ----------------
extern "C" void kernel_launch(void* const* d_in, const int* in_sizes, int n_in,
                              void* d_out, int out_size, void* d_ws, size_t ws_size,
                              hipStream_t stream) {
    const float* inputs = (const float*)d_in[0];
    const float* fm     = (const float*)d_in[1];
    const float* meta   = (const float*)d_in[2];
    const float* mcw    = (const float*)d_in[3];
    const float* mcb    = (const float*)d_in[4];
    const float* mlw    = (const float*)d_in[5];
    const float* mlb    = (const float*)d_in[6];
    const float* icw    = (const float*)d_in[7];
    const float* icb    = (const float*)d_in[8];
    const float* bbw    = (const float*)d_in[9];
    const float* bbb    = (const float*)d_in[10];
    float* out = (float*)d_out;

    const long planeElems = (long)BATCH * 4 * HH * WW;    // 18,874,368
    char* ws = (char*)d_ws;
    __bf16* ci     = (__bf16*)ws;                         // 37,748,736 B
    __bf16* mapped = (__bf16*)(ws + 37748736);            // 37,748,736 B
    __bf16* wts    = (__bf16*)(ws + 75497472);            // 1,048,576 B
    __bf16* wbb    = (__bf16*)(ws + 76546048);            // 32,768 B

    const int nPlaneBlocks = (int)(planeElems / 256);     // 73,728

    zero_bf16_kernel<<<nPlaneBlocks, 256, 0, stream>>>(mapped, (int)planeElems);
    zero_f32_kernel<<<(out_size + 255) / 256, 256, 0, stream>>>(out, out_size);

    cvt_f32_bf16_kernel<<<64, 256, 0, stream>>>(bbw, wbb, 16 * 1024);

    conv3x3_sig_kernel<<<nPlaneBlocks, 256, 0, stream>>>(inputs, icw, icb, ci);

    size_t dynLds = (16 * 1024 + 16 * 1024 + 16 * 256 + 16 * 64) * sizeof(float); // 151,552
    dynw_kernel<<<BATCH, 256, dynLds, stream>>>(fm, meta, mcw, mcb, mlw, mlb, wts);

    size_t convLds = (16 * 1024 + 4 * NROWS * ROWLEN) * sizeof(__bf16);           // 149,248
    dim3 grid(BATCH, NGROUP);
    bigconv_wmma_kernel<<<grid, 256, convLds, stream>>>(
        ci, wts, bbb /*unused*/, mapped, nullptr, 16 * 1024, 0);
    bigconv_wmma_kernel<<<grid, 256, convLds, stream>>>(
        mapped, wbb, bbb, nullptr, out, 0, 1);
}
